// GCNClassifier_14774687498495
// MI455X (gfx1250) — matrix operand order repaired and checked by reading, not compiled, not run on hardware
//
#include <hip/hip_runtime.h>
#include <hip/hip_bf16.h>

// ---- static problem config (mirrors reference) ----
#define B_SZ    64
#define SEQ     337
#define NS      20
#define N_NODES (B_SZ * SEQ * NS)   // 431360 (divisible by 16)
#define N_EDGES (N_NODES * 8)       // 3450880
#define F_IN    3
#define GCN_OUT 29
#define FPAD    32
#define M_ROWS  (B_SZ * NS)         // 1280
#define NCLS    2
#define ALPHA   0.3f

typedef __attribute__((ext_vector_type(16))) _Float16 v16h;
typedef __attribute__((ext_vector_type(8)))  _Float16 v8h;
typedef __attribute__((ext_vector_type(8)))  float    v8f;
typedef __attribute__((ext_vector_type(2)))  float    v2f;

// ============================================================
// 0) bulk zero (16B granules) — clears agg/deg + all padded activations
// ============================================================
__global__ void zero16(uint4* __restrict__ p, long n)
{
    long i = (long)blockIdx.x * blockDim.x + threadIdx.x;
    if (i < n) p[i] = make_uint4(0u, 0u, 0u, 0u);
}

// ============================================================
// 1a) pack GCN weights/bias into padded, branchless form
//     wPad[4][32] (K=3 valid, 29 cols valid), bPad[32]
// ============================================================
__global__ void pack_gcn(const float* __restrict__ w, const float* __restrict__ b,
                         float* __restrict__ wPad, float* __restrict__ bPad)
{
    int i = blockIdx.x * blockDim.x + threadIdx.x;
    if (i < 128) {
        int k = i >> 5, c = i & 31;
        wPad[i] = (k < F_IN && c < GCN_OUT) ? w[k * GCN_OUT + c] : 0.0f;
    } else if (i < 160) {
        int c = i - 128;
        bPad[c] = (c < GCN_OUT) ? b[c] : 0.0f;
    }
}

// ============================================================
// 1b) GCN GEMM: h[N,32] = x[N,3] @ w[3,29] + b  (fp32 WMMA 16x16x4)
//     branchless; 8 waves/block, one 16x16 tile per wave
// ============================================================
__global__ __launch_bounds__(256)
void gcn_gemm_wmma(const float* __restrict__ x, const float* __restrict__ wPad,
                   const float* __restrict__ bPad, float* __restrict__ h)
{
    const int wid  = threadIdx.x >> 5;
    const int lane = threadIdx.x & 31;
    const int tile = blockIdx.x * 8 + wid;        // (N/16)*2 tiles total
    const int tileM = tile >> 1;
    const int tileN = tile & 1;
    const bool hiK = lane >= 16;
    const int  row = tileM * 16 + (lane & 15);
    const int  col = tileN * 16 + (lane & 15);

    // A 16x4: lanes 0-15 {K0,K1}, lanes 16-31 {K2, 0}
    float x0 = x[row * 3 + (hiK ? 2 : 0)];
    float x1 = x[row * 3 + 1];
    v2f a; a.x = x0; a.y = hiK ? 0.0f : x1;
    // B 4x16 from zero-padded wPad: lanes 0-15 {row0,row1}, lanes 16-31 {row2, 0}
    v2f b; b.x = wPad[(hiK ? 64 : 0) + col]; b.y = hiK ? 0.0f : wPad[32 + col];

    v8f c = {};
    c = __builtin_amdgcn_wmma_f32_16x16x4_f32(false, a, false, b, (short)0, c,
                                              false, false);
    const float bv = bPad[col];                   // 0 for padded cols
    long base = ((long)tileM * 16 + (hiK ? 8 : 0)) * FPAD + col;
    #pragma unroll
    for (int r = 0; r < 8; ++r) h[base + (long)r * FPAD] = c[r] + bv;
}

// ============================================================
// 2) edge scatter, wave-per-edge: lane = channel -> coalesced atomics
// ============================================================
__global__ void edge_scatter(const float* __restrict__ h,
                             const int* __restrict__ src, const int* __restrict__ dst,
                             float* __restrict__ agg, float* __restrict__ deg)
{
    long gid = (long)blockIdx.x * blockDim.x + threadIdx.x;
    long e = gid >> 5;
    int  c = (int)(gid & 31);
    if (e >= N_EDGES) return;
    const int s = src[e], d = dst[e];
    if (c < GCN_OUT)
        atomicAdd(&agg[(long)d * FPAD + c], h[(long)s * FPAD + c]);
    else if (c == GCN_OUT)
        atomicAdd(&deg[d], 1.0f);
}

// ============================================================
// 3) finalize: relu(agg/max(deg,1)) ++ x -> f16 feats, transposed to
//    [B*NS][SEQ(+pad)][32]; thread per (node, channel), coalesced
// ============================================================
__global__ void finalize_feats(const float* __restrict__ agg, const float* __restrict__ deg,
                               const float* __restrict__ x, _Float16* __restrict__ feats,
                               int Ltot)
{
    long i = (long)blockIdx.x * blockDim.x + threadIdx.x;
    if (i >= (long)N_NODES * FPAD) return;
    const int c = (int)(i & 31);
    const int n = (int)(i >> 5);
    const int s  = n % NS;
    const int bt = n / NS;            // b*SEQ + t
    const int t  = bt % SEQ;
    const int bb = bt / SEQ;
    float v;
    if (c < GCN_OUT) {
        float dv = deg[n];
        dv = dv > 1.0f ? dv : 1.0f;
        v = agg[i] / dv;
        v = v > 0.0f ? v : 0.0f;
    } else {
        v = x[n * 3 + (c - GCN_OUT)];
    }
    const long mrow = ((long)(bb * NS + s) * Ltot + (t + 2));
    feats[mrow * FPAD + c] = (_Float16)v;
}

// ============================================================
// 4) pack conv weights into exact per-lane WMMA B-fragment order:
//    element index = ((tc*Kchunks + kk)*32 + lane)*16 + hh
//    K = kk*32 + (hh&7) + (hh>>3)*16 + (lane>=16 ? 8:0), co = tc*16 + (lane&15)
//    padded ci/co/k -> 0
// ============================================================
__global__ void pack_wfrag(const float* __restrict__ w, _Float16* __restrict__ wf,
                           int n, int Kchunks, int cilg, int CIorig, int COorig)
{
    int i = blockIdx.x * blockDim.x + threadIdx.x;
    if (i >= n) return;
    const int hh   = i & 15;
    const int lane = (i >> 4) & 31;
    const int kk   = (i >> 9) % Kchunks;
    const int tc   = i / (Kchunks * 512);
    const int K  = kk * 32 + (hh & 7) + ((hh >> 3) << 4) + ((lane >= 16) ? 8 : 0);
    const int co = tc * 16 + (lane & 15);
    const int k  = K >> cilg;
    const int ci = K & ((1 << cilg) - 1);
    float v = 0.0f;
    if (k < 5 && ci < CIorig && co < COorig)
        v = w[(k * CIorig + ci) * COorig + co];
    wf[i] = (_Float16)v;
}

__global__ void pack_bias(const float* __restrict__ b, float* __restrict__ bp,
                          int CO, int COpad)
{
    int i = blockIdx.x * blockDim.x + threadIdx.x;
    if (i < COpad) bp[i] = (i < CO) ? b[i] : 0.0f;
}

// ============================================================
// 5) conv1d SAME (K=5) via WMMA f16 16x16x32, fused bias+LeakyReLU.
//    Fully unconditional inner loop: activations are zero-padded in
//    time (+2 / +tail) and channels; weights zero-padded in K.
//    8 waves/block, each wave a 16(time) x 16(cout) tile.
// ============================================================
__global__ __launch_bounds__(256)
void conv1d_wmma(const _Float16* __restrict__ in, const _Float16* __restrict__ wfrag,
                 const float* __restrict__ biasPad, _Float16* __restrict__ out,
                 int L, int Ltin, int Ltout, int cilg, int COtiles, int tilesT,
                 int Kchunks)
{
    const int wid  = threadIdx.x >> 5;
    const int lane = threadIdx.x & 31;
    const int tile = blockIdx.x * 8 + wid;
    if (tile >= M_ROWS * tilesT * COtiles) return;   // wave-uniform guard
    const int tc  = tile % COtiles;
    const int tmp = tile / COtiles;
    const int tt  = tmp % tilesT;
    const int m   = tmp / tilesT;
    const bool hi = lane >= 16;
    const int CI  = 1 << cilg;
    const int t0  = tt * 16 + (lane & 15);

    // +2*CI: physical row 2 is logical t=0
    const _Float16* inm = in + ((long)m * Ltin + 2) * CI;
    const _Float16* wf  = wfrag + ((long)tc * Kchunks * 32 + lane) * 16;

    v8f acc = {};
    for (int kk = 0; kk < Kchunks; ++kk) {
        const int r0 = kk * 32 + (hi ? 8 : 0);
        const int r1 = r0 + 16;
        const int ti0 = t0 + (r0 >> cilg) - 2;
        const int ti1 = t0 + (r1 >> cilg) - 2;
        v8h a0 = *(const v8h*)(inm + ti0 * CI + (r0 & (CI - 1)));
        v8h a1 = *(const v8h*)(inm + ti1 * CI + (r1 & (CI - 1)));
        v16h a = __builtin_shufflevector(a0, a1, 0, 1, 2, 3, 4, 5, 6, 7,
                                         8, 9, 10, 11, 12, 13, 14, 15);
        v16h b = *(const v16h*)(wf + (long)kk * 512);
        acc = __builtin_amdgcn_wmma_f32_16x16x32_f16(false, a, false, b, (short)0,
                                                     acc, false, false);
    }

    const int co    = tc * 16 + (lane & 15);
    const int COpad = COtiles * 16;
    const float bv  = biasPad[co];
    #pragma unroll
    for (int r = 0; r < 8; ++r) {
        const int to = tt * 16 + r + (hi ? 8 : 0);
        if (to < L) {
            float v = acc[r] + bv;
            v = (v >= 0.0f) ? v : ALPHA * v;
            out[((long)m * Ltout + to + 2) * COpad + co] = (_Float16)v;
        }
    }
}

// ============================================================
// 6) MaxPooling1D(pool=3, stride=3, valid) on padded layouts
// ============================================================
__global__ void maxpool3(const _Float16* __restrict__ in, _Float16* __restrict__ out,
                         int Lin, int Ltin, int Ltout, int C)
{
    const int Lo = Lin / 3;
    long i = (long)blockIdx.x * blockDim.x + threadIdx.x;
    long total = (long)M_ROWS * Lo * C;
    if (i >= total) return;
    int c  = (int)(i % C);
    long r = i / C;
    int to = (int)(r % Lo);
    int mm = (int)(r / Lo);
    const _Float16* p = in + ((long)mm * Ltin + to * 3 + 2) * C + c;
    float v0 = (float)p[0], v1 = (float)p[C], v2 = (float)p[2 * C];
    float v = fmaxf(v0, fmaxf(v1, v2));
    out[((long)mm * Ltout + to + 2) * C + c] = (_Float16)v;
}

// ============================================================
// 7) head: global-avg-pool over 12 steps, then [64]@[64,2]+b
// ============================================================
__global__ void head_kernel(const _Float16* __restrict__ z, const float* __restrict__ wout,
                            const float* __restrict__ bout, float* __restrict__ out,
                            int Ltin)
{
    int i = blockIdx.x * blockDim.x + threadIdx.x;
    if (i >= M_ROWS * NCLS) return;
    int cls = i % NCLS, m = i / NCLS;
    float acc = 0.0f;
    #pragma unroll 1
    for (int c = 0; c < 64; ++c) {
        float mean = 0.0f;
        #pragma unroll
        for (int t = 0; t < 12; ++t) mean += (float)z[((long)m * Ltin + t + 2) * 64 + c];
        mean *= (1.0f / 12.0f);
        acc += mean * wout[c * NCLS + cls];
    }
    out[i] = acc + bout[cls];
}

// ============================================================
// host-side launch
// ============================================================
extern "C" void kernel_launch(void* const* d_in, const int* in_sizes, int n_in,
                              void* d_out, int out_size, void* d_ws, size_t ws_size,
                              hipStream_t stream)
{
    const float* x     = (const float*)d_in[0];
    const int*   src   = (const int*)d_in[1];
    const int*   dst   = (const int*)d_in[2];
    // d_in[3] sensor_indices: identity contiguous partition (unused)
    const float* w_gcn = (const float*)d_in[4];
    const float* b_gcn = (const float*)d_in[5];
    const float* conv_w[7]; const float* conv_b[7];
    for (int i = 0; i < 7; ++i) { conv_w[i] = (const float*)d_in[6 + i];
                                  conv_b[i] = (const float*)d_in[13 + i]; }
    const float* w_out = (const float*)d_in[20];
    const float* b_out = (const float*)d_in[21];
    float*       out   = (float*)d_out;

    // ---- workspace carve-up (all sizes multiples of 256B, contiguous) ----
    char* ws = (char*)d_ws;
    size_t off = 0;
    auto take = [&](size_t bytes) { char* p = ws + off; off += (bytes + 255) & ~(size_t)255; return p; };
    float*    buf_h   = (float*)take((size_t)N_NODES * FPAD * 4);           // 55.2MB
    char*     zbeg    = ws + off;                                           // zero-span start
    float*    buf_agg = (float*)take((size_t)N_NODES * FPAD * 4);           // 55.2MB
    float*    buf_deg = (float*)take((size_t)N_NODES * 4);                  // 1.7MB
    _Float16* feats   = (_Float16*)take((size_t)M_ROWS * 360 * 32 * 2);     // 29.5MB
    _Float16* z1      = (_Float16*)take((size_t)M_ROWS * 360 * 16 * 2);
    _Float16* z2      = (_Float16*)take((size_t)M_ROWS * 360 * 16 * 2);
    _Float16* p1      = (_Float16*)take((size_t)M_ROWS * 120 * 16 * 2);
    _Float16* z3      = (_Float16*)take((size_t)M_ROWS * 120 * 16 * 2);
    _Float16* z4      = (_Float16*)take((size_t)M_ROWS * 120 * 16 * 2);
    _Float16* p2      = (_Float16*)take((size_t)M_ROWS * 56 * 16 * 2);
    _Float16* z5      = (_Float16*)take((size_t)M_ROWS * 56 * 32 * 2);
    _Float16* z6      = (_Float16*)take((size_t)M_ROWS * 56 * 32 * 2);
    _Float16* p3      = (_Float16*)take((size_t)M_ROWS * 24 * 32 * 2);
    _Float16* z7      = (_Float16*)take((size_t)M_ROWS * 24 * 64 * 2);
    size_t zbytes     = (size_t)((ws + off) - zbeg);
    _Float16* wfbase  = (_Float16*)take((size_t)25600 * 2);
    float*    bpbase  = (float*)take((size_t)192 * 4);
    float*    gcnW    = (float*)take((size_t)128 * 4);
    float*    gcnB    = (float*)take((size_t)32 * 4);

    // ---- per-layer conv config (channels padded to pow2>=16) ----
    static const int cilg[7] = { 5, 4, 4, 4, 4, 5, 5};   // log2(CIpad)
    static const int cior[7] = {32, 8, 8, 16, 16, 32, 32};
    static const int coor[7] = { 8, 8, 16, 16, 32, 32, 64};
    static const int cotl[7] = { 1, 1, 1, 1, 2, 2, 4};   // COpad/16
    static const int kchk[7] = { 5, 3, 3, 3, 3, 5, 5};   // Kpad/32 = 5*CIpad/32 ceil
    static const int Lseq[7] = {337, 337, 112, 112, 37, 37, 12};
    static const int tlsT[7] = {22, 22, 7, 7, 3, 3, 1};
    static const int Ltin[7] = {360, 360, 120, 120, 56, 56, 24};
    static const int Ltot[7] = {360, 360, 120, 120, 56, 56, 24};
    size_t wfoff[7], bpoff[7];
    { size_t o = 0, bo = 0;
      for (int i = 0; i < 7; ++i) { wfoff[i] = o; o += (size_t)cotl[i] * kchk[i] * 512;
                                    bpoff[i] = bo; bo += (size_t)cotl[i] * 16; } }

    // 0) zero agg/deg + all padded activation buffers (graph-replay safe)
    {
        long n16 = (long)(zbytes / 16);
        zero16<<<(int)((n16 + 255) / 256), 256, 0, stream>>>((uint4*)zbeg, n16);
    }
    // 1) pack + GCN GEMM (fp32 WMMA 16x16x4)
    pack_gcn<<<1, 160, 0, stream>>>(w_gcn, b_gcn, gcnW, gcnB);
    gcn_gemm_wmma<<<(N_NODES / 16) * 2 / 8, 256, 0, stream>>>(x, gcnW, gcnB, buf_h);
    // 2) edge scatter (wave per edge)
    {
        long thr = (long)N_EDGES * 32;
        edge_scatter<<<(int)((thr + 255) / 256), 256, 0, stream>>>(buf_h, src, dst,
                                                                   buf_agg, buf_deg);
    }
    // 3) relu/concat/transpose -> f16 padded feats
    {
        long thr = (long)N_NODES * FPAD;
        finalize_feats<<<(int)((thr + 255) / 256), 256, 0, stream>>>(buf_agg, buf_deg,
                                                                     x, feats, 360);
    }
    // 4) pack conv weights (B-fragment order) + biases
    for (int i = 0; i < 7; ++i) {
        int n = cotl[i] * kchk[i] * 512;
        pack_wfrag<<<(n + 255) / 256, 256, 0, stream>>>(conv_w[i], wfbase + wfoff[i], n,
                                                        kchk[i], cilg[i], cior[i], coor[i]);
        pack_bias<<<1, cotl[i] * 16, 0, stream>>>(conv_b[i], bpbase + bpoff[i],
                                                  coor[i], cotl[i] * 16);
    }
    // 5) conv stack with pooling
    auto conv = [&](int i, const _Float16* in, _Float16* o) {
        int tiles = M_ROWS * tlsT[i] * cotl[i];
        conv1d_wmma<<<tiles / 8, 256, 0, stream>>>(in, wfbase + wfoff[i], bpbase + bpoff[i],
                                                   o, Lseq[i], Ltin[i], Ltot[i], cilg[i],
                                                   cotl[i], tlsT[i], kchk[i]);
    };
    auto pool = [&](const _Float16* in, _Float16* o, int Lin, int Lti, int Lto, int C) {
        long tot = (long)M_ROWS * (Lin / 3) * C;
        maxpool3<<<(int)((tot + 255) / 256), 256, 0, stream>>>(in, o, Lin, Lti, Lto, C);
    };
    conv(0, feats, z1);              // [1280,337,16(8v)]
    conv(1, z1, z2);                 // [1280,337,16(8v)]
    pool(z2, p1, 337, 360, 120, 16); // [1280,112,16]
    conv(2, p1, z3);                 // [1280,112,16]
    conv(3, z3, z4);                 // [1280,112,16]
    pool(z4, p2, 112, 120, 56, 16);  // [1280,37,16]
    conv(4, p2, z5);                 // [1280,37,32]
    conv(5, z5, z6);                 // [1280,37,32]
    pool(z6, p3, 37, 56, 24, 32);    // [1280,12,32]
    conv(6, p3, z7);                 // [1280,12,64]
    // 6) head
    head_kernel<<<(M_ROWS * NCLS + 255) / 256, 256, 0, stream>>>(z7, w_out, b_out, out, 24);
    (void)in_sizes; (void)n_in; (void)out_size; (void)ws_size;
}
